// CLEPR_17961553231970
// MI455X (gfx1250) — compile-verified, matrix operand order checked by reading, not run on hardware
//
#include <hip/hip_runtime.h>
#include <hip/hip_bf16.h>

#define N_USERS 100000
#define N_ITEMS 30000
#define NTOT    130000
#define NNZ_ADJ  4000000
#define NNZ_SYM  1600000
#define NNZ_HERB 480000

typedef __attribute__((ext_vector_type(16))) __bf16 v16bf;
typedef __attribute__((ext_vector_type(8)))  float  v8f;

#define WMMA_BF16(a,b,c) __builtin_amdgcn_wmma_f32_16x16x32_bf16(false,(a),false,(b),(short)0,(c),false,false)

__device__ __forceinline__ __bf16 f2bf(float f) {
  return (__bf16)f;   // fptrunc f32->bf16 (RNE); native cvt on gfx1250
}

// Branch-free fast tanh on native transcendentals:
// tanh(x) = sign(x) * (1 - 2/(exp(2|x|)+1)); exp2 overflow -> rcp -> 0 -> +/-1.
__device__ __forceinline__ float fast_tanh(float x) {
  const float ax = __builtin_fabsf(x);
  const float e  = __builtin_amdgcn_exp2f(ax * 2.88539008177792681472f); // 2*log2(e)
  const float t  = 1.0f - 2.0f * __builtin_amdgcn_rcpf(1.0f + e);
  return __builtin_copysignf(t, x);
}

// A-operand (16x32 bf16) layout: lane holds row m=lane%16; VGPR v, pos holds
// k = kb*32 + (v>=4 ? 16 : 0) + (lane/16)*8 + (v%4)*2 + pos
__device__ __forceinline__ v16bf load_a_f32(const float* __restrict__ base, int m, int half, int kb) {
  v16bf a;
#pragma unroll
  for (int i = 0; i < 16; ++i) {
    const int v = i >> 1, pos = i & 1;
    const int k = kb * 32 + ((v & 4) ? 16 : 0) + half * 8 + (v & 3) * 2 + pos;
    a[i] = f2bf(base[m * 64 + k]);
  }
  return a;
}

__device__ __forceinline__ v16bf load_b_tile(const unsigned short* __restrict__ p, int tile, int lane) {
  return *(const v16bf*)((const __bf16*)p + (size_t)tile * 512 + lane * 16);
}

// ---------------------------------------------------------------------------
// Pack an f32 (K x 64) matrix into B-operand bf16 tiles:
// out[tile=(kb*4+nb)][lane][i] with n = nb*16 + lane%16, k mirroring A layout.
// ---------------------------------------------------------------------------
__global__ void __launch_bounds__(256)
pack_b_kernel(const float* __restrict__ B, unsigned short* __restrict__ out, int K) {
  const int t = blockIdx.x * blockDim.x + threadIdx.x;
  const int ntiles = (K >> 5) * 4;
  if (t >= ntiles * 32) return;
  const int lane = t & 31, tile = t >> 5;
  const int nb = tile & 3, kb = tile >> 2;
  const int half = lane >> 4, n = lane & 15;
  __bf16* o = (__bf16*)out + (size_t)tile * 512 + lane * 16;
#pragma unroll
  for (int i = 0; i < 16; ++i) {
    const int v = i >> 1, pos = i & 1;
    const int k = kb * 32 + ((v & 4) ? 16 : 0) + half * 8 + (v & 3) * 2 + pos;
    o[i] = f2bf(B[k * 64 + nb * 16 + n]);
  }
}

// ---------------------------------------------------------------------------
// Edge-parallel SpMM: y[rows[e]] += vals[e] * x[cols[e]]   (D=64)
// 16 threads per edge, float4 per thread. x/y stay L2-resident (33 MB < 192 MB).
// ---------------------------------------------------------------------------
__global__ void __launch_bounds__(256)
spmm_kernel(const int* __restrict__ rows, const int* __restrict__ cols,
            const float* __restrict__ vals, const float* __restrict__ x,
            float* __restrict__ y, int nnz) {
  const long long t = (long long)blockIdx.x * blockDim.x + threadIdx.x;
  const int e  = (int)(t >> 4);
  const int c4 = ((int)t & 15) * 4;
  if (e >= nnz) return;
  if (c4 == 0) {
    __builtin_prefetch(rows + e + 2048, 0, 1);
    __builtin_prefetch(cols + e + 2048, 0, 1);
    __builtin_prefetch(vals + e + 2048, 0, 1);
  }
  const int r = rows[e], c = cols[e];
  const float v = vals[e];
  const float4 xv = *(const float4*)(x + (long long)c * 64 + c4);
  float* yp = y + (long long)r * 64 + c4;
  atomicAdd(yp + 0, v * xv.x);
  atomicAdd(yp + 1, v * xv.y);
  atomicAdd(yp + 2, v * xv.z);
  atomicAdd(yp + 3, v * xv.w);
}

// ---------------------------------------------------------------------------
// Fused GraphSAGE layer, one wave per 16-row strip:
//   h   = tanh(agg @ Q)                      (16x64 @ 64x64, 8 WMMAs)
//   ego = tanh([ego, h] @ W + b)             (16x128 @ 128x64, 16 WMMAs)
// h is staged through LDS to convert C-layout -> A-layout.
// ---------------------------------------------------------------------------
__global__ void __launch_bounds__(128)
fused_layer_kernel(const float* __restrict__ agg, float* __restrict__ ego,
                   const unsigned short* __restrict__ Qp,
                   const unsigned short* __restrict__ Wp,
                   const float* __restrict__ bias, int nstrips) {
  __shared__ float hlds[4][16 * 64];
  const int wave = threadIdx.x >> 5;
  const int lane = threadIdx.x & 31;
  const int strip = blockIdx.x * 4 + wave;
  if (strip >= nstrips) return;
  const int half = lane >> 4;
  const int m = lane & 15;          // A row / C column index
  const long long row0 = (long long)strip * 16;
  const float* aggrow = agg + row0 * 64;
  float*       egorow = ego + row0 * 64;

  const v16bf aA0 = load_a_f32(aggrow, m, half, 0);
  const v16bf aA1 = load_a_f32(aggrow, m, half, 1);
  const v16bf aE0 = load_a_f32(egorow, m, half, 0);
  const v16bf aE1 = load_a_f32(egorow, m, half, 1);

  // h = tanh(agg @ Q), write to LDS in row-major f32
#pragma unroll
  for (int nb = 0; nb < 4; ++nb) {
    v8f c = {};
    c = WMMA_BF16(aA0, load_b_tile(Qp, 0 * 4 + nb, lane), c);
    c = WMMA_BF16(aA1, load_b_tile(Qp, 1 * 4 + nb, lane), c);
#pragma unroll
    for (int v = 0; v < 8; ++v)
      hlds[wave][(v + 8 * half) * 64 + nb * 16 + m] = fast_tanh(c[v]);
  }

  // Re-read h in A layout (same wave: LDS in-order; compiler inserts dscnt waits)
  v16bf aH0, aH1;
#pragma unroll
  for (int i = 0; i < 16; ++i) {
    const int v = i >> 1, pos = i & 1;
    const int k = ((v & 4) ? 16 : 0) + half * 8 + (v & 3) * 2 + pos;
    aH0[i] = f2bf(hlds[wave][m * 64 + k]);
    aH1[i] = f2bf(hlds[wave][m * 64 + 32 + k]);
  }

  // ego = tanh([ego, h] @ W + b)   (in-place: wave owns its 16 rows)
#pragma unroll
  for (int nb = 0; nb < 4; ++nb) {
    v8f c = {};
    c = WMMA_BF16(aE0, load_b_tile(Wp, 0 * 4 + nb, lane), c);
    c = WMMA_BF16(aE1, load_b_tile(Wp, 1 * 4 + nb, lane), c);
    c = WMMA_BF16(aH0, load_b_tile(Wp, 2 * 4 + nb, lane), c);
    c = WMMA_BF16(aH1, load_b_tile(Wp, 3 * 4 + nb, lane), c);
    const float bb = bias[nb * 16 + m];
#pragma unroll
    for (int v = 0; v < 8; ++v)
      egorow[(v + 8 * half) * 64 + nb * 16 + m] = fast_tanh(c[v] + bb);
  }
}

// ---------------------------------------------------------------------------
// Pair term: out[row] += tanh(aggp @ M), one wave per 16-row strip.
// ---------------------------------------------------------------------------
__global__ void __launch_bounds__(128)
pair_gemm_add_kernel(const float* __restrict__ aggp,
                     const unsigned short* __restrict__ Mp,
                     float* __restrict__ outp, int nstrips) {
  const int wave = threadIdx.x >> 5;
  const int lane = threadIdx.x & 31;
  const int strip = blockIdx.x * 4 + wave;
  if (strip >= nstrips) return;
  const int half = lane >> 4, m = lane & 15;
  const long long row0 = (long long)strip * 16;
  const float* ar = aggp + row0 * 64;
  float* orow = outp + row0 * 64;
  const v16bf a0 = load_a_f32(ar, m, half, 0);
  const v16bf a1 = load_a_f32(ar, m, half, 1);
#pragma unroll
  for (int nb = 0; nb < 4; ++nb) {
    v8f c = {};
    c = WMMA_BF16(a0, load_b_tile(Mp, 0 * 4 + nb, lane), c);
    c = WMMA_BF16(a1, load_b_tile(Mp, 1 * 4 + nb, lane), c);
#pragma unroll
    for (int v = 0; v < 8; ++v) {
      float* p = orow + (v + 8 * half) * 64 + nb * 16 + m;
      *p = *p + fast_tanh(c[v]);
    }
  }
}

// ---------------------------------------------------------------------------
// Row L2-normalize ego rows [row_start, row_start+row_count) -> d_out (same rows)
// One wave (32 lanes x 2 elems) per row; butterfly reduce over wave32.
// ---------------------------------------------------------------------------
__global__ void __launch_bounds__(256)
normalize_kernel(const float* __restrict__ ego, float* __restrict__ outp,
                 long long row_start, long long row_count) {
  const int wave = threadIdx.x >> 5;
  const int lane = threadIdx.x & 31;
  const long long r = (long long)blockIdx.x * 8 + wave;
  if (r >= row_count) return;
  const float* x = ego + (row_start + r) * 64;
  const float a = x[lane], b = x[lane + 32];
  float s = a * a + b * b;
#pragma unroll
  for (int off = 16; off; off >>= 1) s += __shfl_xor(s, off, 32);
  const float inv = 1.0f / fmaxf(sqrtf(s), 1e-12f);
  float* o = outp + (row_start + r) * 64;
  o[lane]      = a * inv;
  o[lane + 32] = b * inv;
}

extern "C" void kernel_launch(void* const* d_in, const int* in_sizes, int n_in,
                              void* d_out, int out_size, void* d_ws, size_t ws_size,
                              hipStream_t stream) {
  const float* user_emb = (const float*)d_in[0];
  const float* item_emb = (const float*)d_in[1];
  const float* adj_vals = (const float*)d_in[2];
  const float* sym_vals = (const float*)d_in[3];
  const float* herb_vals = (const float*)d_in[4];
  const float* Q_user = (const float*)d_in[5];
  const float* Q_item = (const float*)d_in[6];
  const float* W_user = (const float*)d_in[7];
  const float* W_item = (const float*)d_in[8];
  const float* b_user = (const float*)d_in[9];
  const float* b_item = (const float*)d_in[10];
  const float* M_user = (const float*)d_in[11];
  const float* M_item = (const float*)d_in[12];
  const int* adj_rows = (const int*)d_in[13];
  const int* adj_cols = (const int*)d_in[14];
  const int* sym_rows = (const int*)d_in[15];
  const int* sym_cols = (const int*)d_in[16];
  const int* herb_rows = (const int*)d_in[17];
  const int* herb_cols = (const int*)d_in[18];
  float* out = (float*)d_out;

  const size_t MAT_BYTES = (size_t)NTOT * 64 * sizeof(float);     // 33,280,000 (256-aligned)
  char* ws = (char*)d_ws;
  float* agg = (float*)ws;
  float* ego = (float*)(ws + MAT_BYTES);
  unsigned short* packed = (unsigned short*)(ws + 2 * MAT_BYTES);
  unsigned short* Qu = packed;            // 2 layers x 4096 bf16 (64x64 each)
  unsigned short* Qi = Qu + 2 * 4096;
  unsigned short* Wu = Qi + 2 * 4096;     // 2 layers x 8192 bf16 (128x64 each)
  unsigned short* Wi = Wu + 2 * 8192;
  unsigned short* Mu = Wi + 2 * 8192;     // 4096 bf16
  unsigned short* Mi = Mu + 4096;

  // Pack all weight matrices into WMMA B-operand bf16 tiles (tiny, every call).
  for (int k = 0; k < 2; ++k) {
    pack_b_kernel<<<1, 256, 0, stream>>>(Q_user + k * 4096, Qu + k * 4096, 64);
    pack_b_kernel<<<1, 256, 0, stream>>>(Q_item + k * 4096, Qi + k * 4096, 64);
    pack_b_kernel<<<2, 256, 0, stream>>>(W_user + k * 8192, Wu + k * 8192, 128);
    pack_b_kernel<<<2, 256, 0, stream>>>(W_item + k * 8192, Wi + k * 8192, 128);
  }
  pack_b_kernel<<<1, 256, 0, stream>>>(M_user, Mu, 64);
  pack_b_kernel<<<1, 256, 0, stream>>>(M_item, Mi, 64);

  const int nstrips = NTOT / 16;                          // 8125
  const int adj_blocks = (int)(((long long)NNZ_ADJ * 16 + 255) / 256);

  for (int br = 0; br < 2; ++br) {
    // ego0 = concat(user_embedding, item_embedding)
    hipMemcpyAsync(ego, user_emb, (size_t)N_USERS * 64 * sizeof(float),
                   hipMemcpyDeviceToDevice, stream);
    hipMemcpyAsync(ego + (size_t)N_USERS * 64, item_emb,
                   (size_t)N_ITEMS * 64 * sizeof(float),
                   hipMemcpyDeviceToDevice, stream);
    const unsigned short* Qp = (br == 0) ? Qu : Qi;
    const unsigned short* Wp = (br == 0) ? Wu : Wi;
    const float* bp = (br == 0) ? b_user : b_item;
    for (int k = 0; k < 2; ++k) {
      hipMemsetAsync(agg, 0, MAT_BYTES, stream);
      spmm_kernel<<<adj_blocks, 256, 0, stream>>>(adj_rows, adj_cols, adj_vals,
                                                  ego, agg, NNZ_ADJ);
      fused_layer_kernel<<<(nstrips + 3) / 4, 128, 0, stream>>>(
          agg, ego, Qp + k * 4096, Wp + k * 8192, bp + k * 64, nstrips);
    }
    if (br == 0)
      normalize_kernel<<<N_USERS / 8, 256, 0, stream>>>(ego, out, 0, N_USERS);
    else
      normalize_kernel<<<N_ITEMS / 8, 256, 0, stream>>>(ego, out, N_USERS, N_ITEMS);
  }

  // user_pair: out[:N_USERS] += tanh(spmm(sym, user_emb) @ M_user)
  hipMemsetAsync(agg, 0, (size_t)N_USERS * 64 * sizeof(float), stream);
  spmm_kernel<<<(int)(((long long)NNZ_SYM * 16 + 255) / 256), 256, 0, stream>>>(
      sym_rows, sym_cols, sym_vals, user_emb, agg, NNZ_SYM);
  pair_gemm_add_kernel<<<(N_USERS / 16 + 3) / 4, 128, 0, stream>>>(
      agg, Mu, out, N_USERS / 16);

  // item_pair: out[N_USERS:] += tanh(spmm(herb, item_emb) @ M_item)
  hipMemsetAsync(agg, 0, (size_t)N_ITEMS * 64 * sizeof(float), stream);
  spmm_kernel<<<(int)(((long long)NNZ_HERB * 16 + 255) / 256), 256, 0, stream>>>(
      herb_rows, herb_cols, herb_vals, item_emb, agg, NNZ_HERB);
  pair_gemm_add_kernel<<<(N_ITEMS / 16 + 3) / 4, 128, 0, stream>>>(
      agg, Mi, out + (size_t)N_USERS * 64, N_ITEMS / 16);
}
